// CPC_81905026335197
// MI455X (gfx1250) — matrix operand order repaired and checked by reading, not compiled, not run on hardware
//
#include <hip/hip_runtime.h>

typedef __bf16 bf16;
typedef __attribute__((ext_vector_type(16))) __bf16 v16bf;
typedef __attribute__((ext_vector_type(8)))  __bf16 v8bf;
typedef __attribute__((ext_vector_type(8)))  float  v8f;
typedef __attribute__((ext_vector_type(4)))  unsigned int v4u;
typedef __attribute__((ext_vector_type(8)))  int v8i;
typedef __attribute__((ext_vector_type(4)))  int v4i;

#define T_STEPS 4096
#define HPAD   264   // 256 + 8 bf16 pad -> 528B row, conflict-free 16-row access
#define WFPAD  40    // 32 + 8 pad
#define XPAD   40    // 32 + 8 pad (TDM pad feature produces this)
#define CHUNK  4     // timesteps per TDM transfer (double buffered)

__device__ __forceinline__ v8f splat8(float x) {
  v8f v;
#pragma unroll
  for (int i = 0; i < 8; ++i) v[i] = x;
  return v;
}

__device__ __forceinline__ v16bf ld16(const bf16* p, int off) {
  v8bf lo = *(const v8bf*)p;
  v8bf hi = *(const v8bf*)(p + off);
  return __builtin_shufflevector(lo, hi, 0,1,2,3,4,5,6,7,8,9,10,11,12,13,14,15);
}
#define LDA(p) ld16((p), 16)   // A-operand: chunks at +0 and +16 elements
#define LDB(p) ld16((p), 16)   // B-operand: chunks at +0 and +16 elements

__device__ __forceinline__ v8f wmma_bf16(v16bf a, v16bf b, v8f c) {
  return __builtin_amdgcn_wmma_f32_16x16x32_bf16(false, a, false, b, (short)0, c, false, false);
}

__device__ __forceinline__ float fsigmoid(float x) {
  float e = __expf(-x);
  return __builtin_amdgcn_rcpf(1.0f + e);
}
__device__ __forceinline__ float ftanh_f(float x) {
  float ax = __builtin_fabsf(x);
  float e  = __expf(2.0f * ax);
  float r  = 1.0f - 2.0f * __builtin_amdgcn_rcpf(1.0f + e);
  return __builtin_copysignf(r, x);
}

// Issue one TDM transfer: CHUNK*2048 bf16 from global -> LDS, padding each
// 64B row (32 bf16) with 16B so LDS rows land on XPAD=40-element strides.
__device__ __forceinline__ void tdm_issue(const bf16* gsrc, unsigned ldsoff) {
  unsigned long long ga = (unsigned long long)(uintptr_t)gsrc;
  v4u g0;
  g0[0] = 1u;                                                  // count=1
  g0[1] = ldsoff;                                              // lds_addr
  g0[2] = (unsigned)ga;                                        // global_addr lo
  g0[3] = (unsigned)((ga >> 32) & 0x1FFFFFFull) | (2u << 30);  // addr hi | type=2
  v8i g1;
  g1[0] = (int)((1u << 16) | (1u << 20) | (3u << 22) | (3u << 25));
  //        data_size=2B | pad_enable | pad_interval=16dw(64B) | pad_amount=4dw(16B)
  g1[1] = (int)(8192u << 16);   // tensor_dim0 = 8192 elements
  g1[2] = (int)(1u << 16);      // tensor_dim1 = 1
  g1[3] = (int)(8192u << 16);   // tile_dim0 = 8192
  g1[4] = 0;                    // tile_dim1 = tile_dim2 = 0 (1D tile)
  g1[5] = 8192;                 // tensor_dim0_stride
  g1[6] = 0;
  g1[7] = 0;
  v4i z4 = {0, 0, 0, 0};
  v8i z8 = {0, 0, 0, 0, 0, 0, 0, 0};
  __builtin_amdgcn_tensor_load_to_lds(g0, g1, z4, z4, z8, 0);
}

// ---------------- prep: fused weights + bf16 conversion + sync reset ----------------
__global__ __launch_bounds__(256) void cpc_prep_w(
    const float* __restrict__ Wih, const float* __restrict__ Wenc,
    const float* __restrict__ Whh, bf16* __restrict__ Wf,
    bf16* __restrict__ WhhB, unsigned* __restrict__ syncc) {
  int tid = threadIdx.x;
  if (tid < 4) syncc[tid] = 0u;
  // Wf[g][c] = sum_d Wih[g][d] * Wenc[d][c]   (768 x 32)
  for (int idx = tid; idx < 768 * 32; idx += 256) {
    int g = idx >> 5, c = idx & 31;
    float s = 0.0f;
    for (int d = 0; d < 128; ++d) s = fmaf(Wih[g * 128 + d], Wenc[d * 32 + c], s);
    Wf[idx] = (bf16)s;
  }
  for (int idx = tid; idx < 768 * 256; idx += 256) WhhB[idx] = (bf16)Whh[idx];
}

// ---------------- prep: batch (B,C,T) f32 -> batchT (T,B,C) bf16 ----------------
__global__ __launch_bounds__(256) void cpc_prep_x(
    const float* __restrict__ batch, bf16* __restrict__ batchT) {
  int t = blockIdx.x;
  int base = threadIdx.x * 8;
  v8bf o;
#pragma unroll
  for (int q = 0; q < 8; ++q) {
    int idx = base + q;
    int b = idx >> 5, c = idx & 31;
    o[q] = (bf16)batch[(size_t)(b * 32 + c) * 4096 + t];
  }
  *(v8bf*)(batchT + (size_t)t * 2048 + base) = o;
}

// ---------------- GRU scan: 2 blocks (one WGP each), channel-split ----------------
__global__ __launch_bounds__(256, 1) void cpc_gru(
    const bf16* __restrict__ WhhB, const bf16* __restrict__ WfB,
    const bf16* __restrict__ batchT, const float* __restrict__ bih,
    const float* __restrict__ bhh, const float* __restrict__ h0,
    const int* __restrict__ tpos, float* __restrict__ ct,
    bf16* __restrict__ hx, unsigned* __restrict__ syncc) {
  extern __shared__ char sm[];
  bf16* sWhh = (bf16*)sm;                 // [384][HPAD]  bf16  (r/z/n x 128ch, K=256)
  bf16* sWf  = sWhh + 384 * HPAD;         // [384][WFPAD] bf16  (fused input weights, K=32)
  bf16* sH   = sWf  + 384 * WFPAD;        // [64][HPAD]   bf16  (full hidden state)
  bf16* sX   = sH   + 64  * HPAD;         // [2][CHUNK][64][XPAD] bf16 (double-buffered)

  const int tid = threadIdx.x;
  const int lid = tid & 31;
  const int w   = tid >> 5;        // wave 0..7 -> owns channels [16w,16w+16)
  const int wg  = blockIdx.x;      // 0/1 -> channel halves
  const int hs  = wg * 128;
  const int hi  = lid >> 4;
  const int nl  = lid & 15;
  const int C   = hs + w * 16 + nl;   // global channel owned by this lane

  // load W_hh slice: local row j -> gate g=j>>7, channel hs+(j&127)
  for (int j = w; j < 384; j += 8) {
    int g = j >> 7, chn = j & 127;
    int grow = g * 256 + hs + chn;
    ((uint4*)(sWhh + j * HPAD))[lid] = ((const uint4*)(WhhB + (size_t)grow * 256))[lid];
  }
  // load fused input-weight slice
  {
    int rr = tid >> 2, cc = tid & 3;
    for (int j = rr; j < 384; j += 64) {
      int g = j >> 7, chn = j & 127;
      int grow = g * 256 + hs + chn;
      ((uint4*)(sWf + j * WFPAD))[cc] = ((const uint4*)(WfB + (size_t)grow * 32))[cc];
    }
  }
  // init hidden state from h0
  for (int idx = tid; idx < 64 * 256; idx += 256) {
    int m = idx >> 8, c = idx & 255;
    sH[m * HPAD + c] = (bf16)h0[idx];
  }

  const float bias_r  = bih[C]       + bhh[C];
  const float bias_z  = bih[256 + C] + bhh[256 + C];
  const float bias_in = bih[512 + C];
  const float bias_hn = bhh[512 + C];

  // t_pos for the 32 rows this lane produces, packed 2x16-bit per VGPR
  unsigned tpmp[16];
#pragma unroll
  for (int mt = 0; mt < 4; ++mt)
#pragma unroll
    for (int i2 = 0; i2 < 4; ++i2) {
      int m0 = mt * 16 + 2 * i2 + hi * 8;
      tpmp[mt * 4 + i2] = (unsigned)tpos[m0] | ((unsigned)tpos[m0 + 1] << 16);
    }

  const unsigned sx_base = (unsigned)(unsigned long long)((const char*)sX - (const char*)sm);
  const unsigned sx_bufbytes = CHUNK * 64 * XPAD * 2;

  // prologue: kick DMA for chunk 0 into buffer 0
  if (tid == 0) tdm_issue(batchT, sx_base);
  __syncthreads();

  unsigned target = 0;
  const v8f zero8 = splat8(0.0f);
  for (int t = 0; t < T_STEPS; ++t) {
    const int tl  = t & (CHUNK - 1);
    const int buf = (t >> 2) & 1;
    if (tl == 0) {
      if (tid == 0) {
        if (t + CHUNK < T_STEPS) {
          // prefetch next chunk into the other buffer, then wait only for the
          // previous transfer (tensor ops complete in order per wave)
          tdm_issue(batchT + (size_t)(t + CHUNK) * 2048, sx_base + (buf ^ 1) * sx_bufbytes);
          __builtin_amdgcn_s_wait_tensorcnt(1);
        } else {
          __builtin_amdgcn_s_wait_tensorcnt(0);
        }
      }
      __syncthreads();
    }

    v8f accR[4], accZ[4], accIN[4], accHN[4];

    // gi = x_t @ Wf^T  (K=32, one WMMA step per tile; C starts as inline 0)
    {
      const bf16* xb = sX + ((buf * CHUNK + tl) * 64) * XPAD;
      v16bf ax[4];
#pragma unroll
      for (int mt = 0; mt < 4; ++mt)
        ax[mt] = LDA(xb + (mt * 16 + nl) * XPAD + hi * 8);
      v16bf bb;
      bb = LDB(sWf + (w * 16 + nl) * WFPAD + hi * 8);
#pragma unroll
      for (int mt = 0; mt < 4; ++mt) accR[mt] = wmma_bf16(ax[mt], bb, zero8);
      bb = LDB(sWf + (128 + w * 16 + nl) * WFPAD + hi * 8);
#pragma unroll
      for (int mt = 0; mt < 4; ++mt) accZ[mt] = wmma_bf16(ax[mt], bb, zero8);
      bb = LDB(sWf + (256 + w * 16 + nl) * WFPAD + hi * 8);
#pragma unroll
      for (int mt = 0; mt < 4; ++mt) accIN[mt] = wmma_bf16(ax[mt], bb, zero8);
    }

    // gh = h @ W_hh^T  (K=256 -> 8 WMMA K-steps)
#pragma unroll
    for (int k0 = 0; k0 < 256; k0 += 32) {
      v16bf ah[4];
#pragma unroll
      for (int mt = 0; mt < 4; ++mt)
        ah[mt] = LDA(sH + (mt * 16 + nl) * HPAD + k0 + hi * 8);
      v16bf bb;
      bb = LDB(sWhh + (w * 16 + nl) * HPAD + k0 + hi * 8);
#pragma unroll
      for (int mt = 0; mt < 4; ++mt) accR[mt] = wmma_bf16(ah[mt], bb, accR[mt]);
      bb = LDB(sWhh + (128 + w * 16 + nl) * HPAD + k0 + hi * 8);
#pragma unroll
      for (int mt = 0; mt < 4; ++mt) accZ[mt] = wmma_bf16(ah[mt], bb, accZ[mt]);
      bb = LDB(sWhh + (256 + w * 16 + nl) * HPAD + k0 + hi * 8);
#pragma unroll
      for (int mt = 0; mt < 4; ++mt) {
        accHN[mt] = (k0 == 0) ? wmma_bf16(ah[mt], bb, zero8)
                              : wmma_bf16(ah[mt], bb, accHN[mt]);
      }
    }

    __syncthreads();  // all reads of sH(t-1) done before anyone writes h(t)

    const int par = t & 1;
    bf16* hxw = hx + (size_t)((par * 2 + wg) * 64) * 128;
#pragma unroll
    for (int mt = 0; mt < 4; ++mt) {
#pragma unroll
      for (int i = 0; i < 8; ++i) {
        int m = mt * 16 + i + hi * 8;
        float r  = fsigmoid(accR[mt][i] + bias_r);
        float u  = fsigmoid(accZ[mt][i] + bias_z);
        float n  = ftanh_f(fmaf(r, accHN[mt][i] + bias_hn, accIN[mt][i] + bias_in));
        float ho = (float)sH[m * HPAD + C];          // previous h (bf16)
        float hv = fmaf(u, ho - n, n);               // (1-u)*n + u*h
        bf16 hb = (bf16)hv;
        sH[m * HPAD + C] = hb;                       // own slice into LDS
        hxw[m * 128 + (w * 16 + nl)] = hb;           // own slice into exchange buf
        unsigned tp = tpmp[mt * 4 + (i >> 1)];
        unsigned tv = (i & 1) ? (tp >> 16) : (tp & 0xFFFFu);
        if ((unsigned)t == tv) ct[m * 256 + C] = hv; // capture c_t at t_pos[m]
      }
    }

    // cross-WGP barrier (monotonic counter through L2)
    target += 2;
    __threadfence();
    __syncthreads();
    if (tid == 0) {
      __hip_atomic_fetch_add(syncc, 1u, __ATOMIC_RELEASE, __HIP_MEMORY_SCOPE_AGENT);
      while (__hip_atomic_load(syncc, __ATOMIC_ACQUIRE, __HIP_MEMORY_SCOPE_AGENT) < target)
        __builtin_amdgcn_s_sleep(1);
    }
    __syncthreads();

    // import partner's half of h(t) into LDS
    {
      const int m  = tid >> 2;
      const int c0 = (tid & 3) * 32;
      const bf16* src = hx + (size_t)((par * 2 + (1 - wg)) * 64 + m) * 128 + c0;
      bf16* dst = sH + m * HPAD + (128 - hs) + c0;
      uint4 v0 = ((const uint4*)src)[0];
      uint4 v1 = ((const uint4*)src)[1];
      uint4 v2 = ((const uint4*)src)[2];
      uint4 v3 = ((const uint4*)src)[3];
      ((uint4*)dst)[0] = v0;
      ((uint4*)dst)[1] = v1;
      ((uint4*)dst)[2] = v2;
      ((uint4*)dst)[3] = v3;
    }
    __syncthreads();
  }
}

// ---------------- tail: pred/enc/total/log-softmax diag, one block per k ----------------
__global__ __launch_bounds__(256, 1) void cpc_final(
    const float* __restrict__ ct, const int* __restrict__ tpos,
    const float* __restrict__ batch, const float* __restrict__ Wenc,
    const float* __restrict__ Wkw, const float* __restrict__ Wkb,
    float* __restrict__ partials) {
  extern __shared__ char sm[];
  float* sct  = (float*)sm;            // [64][256]
  float* pred = sct  + 64 * 256;       // [64][130]
  float* enc  = pred + 64 * 130;       // [64][130]
  float* xk   = enc  + 64 * 130;       // [64][32]   (reused as red[64])
  float* tot  = xk   + 64 * 32;        // [64][66]
  const int k = blockIdx.x, tid = threadIdx.x;

  for (int i = tid; i < 64 * 256 / 4; i += 256) ((float4*)sct)[i] = ((const float4*)ct)[i];
  __syncthreads();

  // pred[c][d] = ct[c] . Wkw[k][d] + Wkb[k][d]
  {
    int d = tid & 127, ch = tid >> 7;
    float acc[32];
#pragma unroll
    for (int i = 0; i < 32; ++i) acc[i] = 0.0f;
    const float* wr = Wkw + ((size_t)k * 128 + d) * 256;
    for (int hh = 0; hh < 256; ++hh) {
      float wv = wr[hh];
#pragma unroll
      for (int cc = 0; cc < 32; ++cc)
        acc[cc] = fmaf(wv, sct[(ch * 32 + cc) * 256 + hh], acc[cc]);
    }
    float bv = Wkb[k * 128 + d];
#pragma unroll
    for (int cc = 0; cc < 32; ++cc) pred[(ch * 32 + cc) * 130 + d] = acc[cc] + bv;
  }
  // stage x at t_pos[b]+1+k
  for (int idx = tid; idx < 2048; idx += 256) {
    int b = idx >> 5, c = idx & 31;
    int ti = tpos[b] + 1 + k;
    xk[idx] = batch[(size_t)(b * 32 + c) * 4096 + ti];
  }
  __syncthreads();
  // enc[b][d] = x . Wenc[d]  (exact f32 encoder at the K needed positions)
  {
    int d = tid & 127, bh = tid >> 7;
    float we[32];
#pragma unroll
    for (int c = 0; c < 32; ++c) we[c] = Wenc[d * 32 + c];
    for (int bb = 0; bb < 32; ++bb) {
      int b = bh * 32 + bb;
      float s = 0.0f;
#pragma unroll
      for (int c = 0; c < 32; ++c) s = fmaf(we[c], xk[b * 32 + c], s);
      enc[b * 130 + d] = s;
    }
  }
  __syncthreads();
  // tot[b][j] = enc[b] . pred[j]
  {
    int j = tid & 63, g = tid >> 6;
    for (int bb = 0; bb < 16; ++bb) {
      int b = g * 16 + bb;
      float s = 0.0f;
      for (int d = 0; d < 128; ++d) s = fmaf(enc[b * 130 + d], pred[j * 130 + d], s);
      tot[b * 66 + j] = s;
    }
  }
  __syncthreads();
  float* red = xk;
  if (tid < 64) {
    int b = tid;
    float mx = -3.4e38f;
    for (int j = 0; j < 64; ++j) mx = fmaxf(mx, tot[b * 66 + j]);
    float s = 0.0f;
    for (int j = 0; j < 64; ++j) s += __expf(tot[b * 66 + j] - mx);
    float lse = mx + __logf(s);
    red[b] = tot[b * 66 + b] - lse;
  }
  __syncthreads();
  if (tid == 0) {
    float s = 0.0f;
    for (int b = 0; b < 64; ++b) s += red[b];
    partials[k] = s;
  }
}

__global__ void cpc_reduce(const float* __restrict__ partials, float* __restrict__ out) {
  if (threadIdx.x == 0) {
    float s = 0.0f;
    for (int i = 0; i < 12; ++i) s += partials[i];
    out[0] = s * (-1.0f / 768.0f);   // / (-B*K)
  }
}

extern "C" void kernel_launch(void* const* d_in, const int* in_sizes, int n_in,
                              void* d_out, int out_size, void* d_ws, size_t ws_size,
                              hipStream_t stream) {
  (void)in_sizes; (void)n_in; (void)out_size; (void)ws_size;
  const float* batch = (const float*)d_in[0];
  const int*   tpos  = (const int*)d_in[1];
  const float* h0    = (const float*)d_in[2];
  const float* Wenc  = (const float*)d_in[3];
  const float* Wih   = (const float*)d_in[4];
  const float* Whh   = (const float*)d_in[5];
  const float* bih   = (const float*)d_in[6];
  const float* bhh   = (const float*)d_in[7];
  const float* Wkw   = (const float*)d_in[8];
  const float* Wkb   = (const float*)d_in[9];

  char* ws = (char*)d_ws;
  bf16*     Wf       = (bf16*)(ws + 0);         //  49152 B
  bf16*     WhhB     = (bf16*)(ws + 49152);     // 393216 B
  float*    ct       = (float*)(ws + 442368);   //  65536 B
  bf16*     hx       = (bf16*)(ws + 507904);    //  65536 B (2 parities x 2 slices)
  unsigned* syncc    = (unsigned*)(ws + 573440);
  float*    partials = (float*)(ws + 573696);
  bf16*     batchT   = (bf16*)(ws + 573952);    // 16 MiB (T,B,C) bf16

  cpc_prep_w<<<1, 256, 0, stream>>>(Wih, Wenc, Whh, Wf, WhhB, syncc);
  cpc_prep_x<<<4096, 256, 0, stream>>>(batch, batchT);

  size_t smem_gru = (size_t)(384 * HPAD + 384 * WFPAD + 64 * HPAD + 2 * CHUNK * 64 * XPAD) * sizeof(bf16);
  cpc_gru<<<2, 256, smem_gru, stream>>>(WhhB, Wf, batchT, bih, bhh, h0, tpos, ct, hx, syncc);

  size_t smem_fin = (size_t)(64 * 256 + 64 * 130 + 64 * 130 + 64 * 32 + 64 * 66) * sizeof(float);
  cpc_final<<<12, 256, smem_fin, stream>>>(ct, tpos, batch, Wenc, Wkw, Wkb, partials);
  cpc_reduce<<<1, 32, 0, stream>>>(partials, (float*)d_out);
}
